// MixConv_34342558499546
// MI455X (gfx1250) — compile-verified, matrix operand order checked by reading, not run on hardware
//
#include <hip/hip_runtime.h>
#include <hip/hip_bf16.h>

// ---------------- problem constants ----------------
#define BSZ   16
#define NPTS  2048
#define NTOTC 32768          // BSZ*NPTS
#define KNN   32
#define NEDGE 524288
#define ME    1048576        // NTOTC*KNN edge rows
#define EPSB  1e-5f

typedef __attribute__((ext_vector_type(16))) _Float16 v16h;
typedef __attribute__((ext_vector_type(8)))  float    v8f;

union Frag16 { uint4 q[2]; v16h v; };

// A/B fragment for V_WMMA_F32_16X16X32_F16 (wave32):
// lane l, element j:  k = (l<16 ? 0 : 8) + (j<8 ? j : j+8)  -> two contiguous runs of 8 halves
__device__ __forceinline__ v16h load_ab(const _Float16* p, int lane) {
  const int boff = (lane & 16) ? 8 : 0;
  Frag16 f;
  f.q[0] = *reinterpret_cast<const uint4*>(p + boff);
  f.q[1] = *reinterpret_cast<const uint4*>(p + boff + 16);
  return f.v;
}

// cooperative stage of a 64(N) x 32(K) f16 panel of Bt into LDS (row-major, 32 halves/row)
__device__ __forceinline__ void stage_b(_Float16* dst, const _Float16* Bb,
                                        int n0, int Kd, int k0, int t) {
  const int r = t >> 1, seg = (t & 1) * 16;          // 128 threads cover 64 rows x 2 segs
  const uint4* s = reinterpret_cast<const uint4*>(Bb + (long long)(n0 + r) * Kd + k0 + seg);
  uint4* d = reinterpret_cast<uint4*>(dst + r * 32 + seg);
  d[0] = s[0];
  d[1] = s[1];
}

// ---------------- generic WMMA GEMM ----------------
// C[M,N](f32) = act( A[M,K]f16 @ Bt[N,K]^T f16 + bias ), optional accumulate,
// optional per-channel sum/sumsq atomics for BatchNorm.
// Block = 128 threads = 4 waves; each wave owns a 16x64 output strip
// (1 M-tile x 4 N-tiles, A fragment reused 4x). B panel double-buffered in LDS.
// Requires M % 64 == 0, N % 64 == 0, K % 32 == 0 (all call sites satisfy this).
// flags: 1=ReLU, 2=accumulate into C
__global__ void __launch_bounds__(128)
k_gemm(const _Float16* __restrict__ A, long long sA,
       const _Float16* __restrict__ Bt, long long sB,
       const float* __restrict__ bias,
       float* __restrict__ C, long long sC,
       int M, int Nn, int Kd, int flags, float* __restrict__ stats)
{
  __shared__ _Float16 lB[2][64 * 32];                // 8 KB double buffer
  const int t    = threadIdx.x;
  const int lane = t & 31;
  const int wave = t >> 5;
  const int m0 = (blockIdx.x * 4 + wave) * 16;
  const int n0 = blockIdx.y * 64;
  if (blockIdx.x * 64 >= M) return;                  // block-uniform (M % 64 == 0)
  const _Float16* Ab = A + (long long)blockIdx.z * sA;
  const _Float16* Bb = Bt + (long long)blockIdx.z * sB;
  float* Cb = C + (long long)blockIdx.z * sC;
  const int lidx = lane & 15;
  const long long arow = (long long)(m0 + lidx) * Kd;

  v8f acc[4];
  #pragma unroll
  for (int nt = 0; nt < 4; ++nt) acc[nt] = (v8f){};

  const int nsteps = Kd >> 5;
  stage_b(lB[0], Bb, n0, Kd, 0, t);
  for (int it = 0; it < nsteps; ++it) {
    __syncthreads();                                 // staged panel visible / safe to reuse
    if (it + 1 < nsteps)                             // overlap next stage with 4 WMMAs
      stage_b(lB[(it + 1) & 1], Bb, n0, Kd, (it + 1) * 32, t);
    const _Float16* pa = Ab + arow + (long long)it * 32;
    if (it + 1 < nsteps) __builtin_prefetch(pa + 32, 0, 1);
    v16h a = load_ab(pa, lane);
    const _Float16* lb = lB[it & 1];
    #pragma unroll
    for (int nt = 0; nt < 4; ++nt) {
      v16h b = load_ab(lb + (nt * 16 + lidx) * 32, lane);   // ds_load_b128 x2
      acc[nt] = __builtin_amdgcn_wmma_f32_16x16x32_f16(false, a, false, b,
                                                       (short)0, acc[nt], false, false);
    }
  }

  const int roff = (lane & 16) ? 8 : 0;
  #pragma unroll
  for (int nt = 0; nt < 4; ++nt) {
    const int col = n0 + nt * 16 + lidx;
    const float bv = bias ? bias[col] : 0.0f;
    float s0 = 0.f, s1 = 0.f;
    #pragma unroll
    for (int r = 0; r < 8; ++r) {
      long long ci = (long long)(m0 + roff + r) * Nn + col;
      float v = acc[nt][r] + bv;
      if (flags & 2) v += Cb[ci];
      if (flags & 1) v = v > 0.f ? v : 0.f;
      Cb[ci] = v;
      s0 += v; s1 += v * v;
    }
    if (stats) { atomicAdd(stats + col, s0); atomicAdd(stats + Nn + col, s1); }
  }
}

// ---------------- weight prep: W[K,N]f32 -> Bt[N,Kpad]f16 ----------------
__global__ void k_wt(const float* __restrict__ W, _Float16* __restrict__ Bt,
                     int K, int N, int Kpad) {
  int i = blockIdx.x * blockDim.x + threadIdx.x;
  if (i >= N * Kpad) return;
  int n = i / Kpad, k = i % Kpad;
  Bt[i] = (k < K) ? (_Float16)W[(long long)k * N + n] : (_Float16)0.f;
}

// ---------------- misc elementwise ----------------
__global__ void k_pos16(const float* __restrict__ x4, _Float16* __restrict__ p) {
  int t = blockIdx.x * blockDim.x + threadIdx.x;
  if (t >= NTOTC) return;
  p[t*3+0] = (_Float16)x4[t*4+0];
  p[t*3+1] = (_Float16)x4[t*4+1];
  p[t*3+2] = (_Float16)x4[t*4+2];
}

__global__ void k_cast_pad(const float* __restrict__ src, int Cs, int C,
                           _Float16* __restrict__ dst, int Cpad, long long total) {
  long long i = (long long)blockIdx.x * blockDim.x + threadIdx.x;
  if (i >= total) return;
  int c = (int)(i % Cpad);
  long long m = i / Cpad;
  dst[i] = (c < C) ? (_Float16)src[m * Cs + c] : (_Float16)0.f;
}

__global__ void k_bn_finalize(float* __restrict__ stats, const float* __restrict__ g,
                              const float* __restrict__ bt, int Nc, float invM) {
  int n = blockIdx.x * blockDim.x + threadIdx.x;
  if (n >= Nc) return;
  float mean = stats[n] * invM;
  float var  = stats[Nc + n] * invM - mean * mean;
  float sc   = g[n] * rsqrtf(var + EPSB);
  stats[n]      = sc;
  stats[Nc + n] = bt[n] - mean * sc;
}

__global__ void k_bn_apply(const float* __restrict__ Y, const float* __restrict__ stats,
                           _Float16* __restrict__ o, long long total, int Nc) {
  long long i = (long long)blockIdx.x * blockDim.x + threadIdx.x;
  if (i >= total) return;
  int c = (int)(i % Nc);
  o[i] = (_Float16)(Y[i] * stats[c] + stats[Nc + c]);
}

__global__ void k_brc(const float* __restrict__ Y, const float* __restrict__ bias,
                      float* __restrict__ w32, _Float16* __restrict__ w16,
                      long long total, int C) {
  long long i = (long long)blockIdx.x * blockDim.x + threadIdx.x;
  if (i >= total) return;
  int c = (int)(i % C);
  float v = Y[i] + bias[c];
  v = v > 0.f ? v : 0.f;
  if (w32) w32[i] = v;
  w16[i] = (_Float16)v;
}

__global__ void k_concat2(const _Float16* __restrict__ a, int Ca,
                          const _Float16* __restrict__ b, int Cb,
                          _Float16* __restrict__ o, long long M) {
  int Ct = Ca + Cb;
  long long total = M * Ct;
  long long i = (long long)blockIdx.x * blockDim.x + threadIdx.x;
  if (i >= total) return;
  int c = (int)(i % Ct);
  long long m = i / Ct;
  o[i] = (c < Ca) ? a[m * Ca + c] : b[m * Cb + (c - Ca)];
}

__global__ void k_max_k(const _Float16* __restrict__ h, _Float16* __restrict__ o, int C) {
  int i = blockIdx.x * blockDim.x + threadIdx.x;
  if (i >= NTOTC * C) return;
  int node = i / C, c = i % C;
  const _Float16* p = h + ((long long)node * KNN) * C + c;
  float m = -3.4e38f;
  for (int kk = 0; kk < KNN; ++kk) {
    float v = (float)p[(long long)kk * C];
    m = v > m ? v : m;
  }
  o[(long long)node * C + c] = (_Float16)m;
}

// ---------------- KNN ----------------
__global__ void k_knn_pos(const float* __restrict__ x4, int* __restrict__ idx) {
  int t = blockIdx.x * blockDim.x + threadIdx.x;
  if (t >= NTOTC) return;
  int b = t >> 11;
  const float* xb = x4 + ((long long)b * NPTS) * 4;
  float px = x4[t*4], py = x4[t*4+1], pz = x4[t*4+2];
  float kd[KNN]; int ki[KNN];
  for (int r = 0; r < KNN; ++r) { kd[r] = 3.4e38f; ki[r] = 0; }
  float mx = 3.4e38f; int mp = 0;
  for (int j = 0; j < NPTS; ++j) {
    float dx = xb[j*4] - px, dy = xb[j*4+1] - py, dz = xb[j*4+2] - pz;
    float d = dx*dx + dy*dy + dz*dz;
    if (d < mx) {
      kd[mp] = d; ki[mp] = j;
      mx = kd[0]; mp = 0;
      for (int r = 1; r < KNN; ++r) if (kd[r] > mx) { mx = kd[r]; mp = r; }
    }
  }
  for (int r = 0; r < KNN; ++r) idx[(long long)t * KNN + r] = ki[r];
}

__global__ void k_norms(const _Float16* __restrict__ x, float* __restrict__ nrm) {
  int t = blockIdx.x * blockDim.x + threadIdx.x;
  if (t >= NTOTC) return;
  float s = 0.f;
  for (int c = 0; c < 64; ++c) { float v = (float)x[(long long)t*64 + c]; s += v*v; }
  nrm[t] = s;
}

__global__ void k_topk_gram(const float* __restrict__ G, const float* __restrict__ nrm,
                            int* __restrict__ idx) {
  int t = blockIdx.x * blockDim.x + threadIdx.x;
  if (t >= NTOTC) return;
  int b = t >> 11, i = t & (NPTS - 1);
  const float* g  = G + (long long)b * NPTS * NPTS + (long long)i * NPTS;
  const float* nb = nrm + (long long)b * NPTS;
  float kd[KNN]; int ki[KNN];
  for (int r = 0; r < KNN; ++r) { kd[r] = 3.4e38f; ki[r] = 0; }
  float mx = 3.4e38f; int mp = 0;
  for (int j = 0; j < NPTS; ++j) {
    float d = nb[j] - 2.0f * g[j];         // row-constant |x_i|^2 dropped (rank-invariant)
    if (d < mx) {
      kd[mp] = d; ki[mp] = j;
      mx = kd[0]; mp = 0;
      for (int r = 1; r < KNN; ++r) if (kd[r] > mx) { mx = kd[r]; mp = r; }
    }
  }
  for (int r = 0; r < KNN; ++r) idx[(long long)t * KNN + r] = ki[r];
}

// ---------------- edge features: [xi, xj - xi] padded ----------------
__global__ void k_edge_features(const _Float16* __restrict__ xn, const int* __restrict__ idx,
                                _Float16* __restrict__ xe, int C, int Cpad) {
  int e = blockIdx.x * blockDim.x + threadIdx.x;
  if (e >= ME) return;
  int node = e >> 5;
  int b = node >> 11;
  int j = idx[e];
  long long jg = ((long long)b << 11) + j;
  const _Float16* xi = xn + (long long)node * C;
  const _Float16* xj = xn + jg * C;
  _Float16* o = xe + (long long)e * Cpad;
  for (int c = 0; c < C; ++c) {
    _Float16 a = xi[c];
    o[c]     = a;
    o[C + c] = (_Float16)((float)xj[c] - (float)a);
  }
  for (int c = 2 * C; c < Cpad; ++c) o[c] = (_Float16)0.f;
}

// ---------------- TAG conv pieces ----------------
__global__ void k_deg(const int* __restrict__ col, float* __restrict__ deg) {
  int e = blockIdx.x * blockDim.x + threadIdx.x;
  if (e >= NEDGE) return;
  atomicAdd(deg + col[e], 1.0f);
}

__global__ void k_dinv(const float* __restrict__ deg, float* __restrict__ dinv) {
  int t = blockIdx.x * blockDim.x + threadIdx.x;
  if (t >= NTOTC) return;
  float d = deg[t];
  dinv[t] = d > 0.f ? rsqrtf(fmaxf(d, 1.f)) : 0.f;
}

__global__ void k_prop(const float* __restrict__ x, const int* __restrict__ row,
                       const int* __restrict__ col, const float* __restrict__ dinv,
                       float* __restrict__ out, int C, int Cs) {
  int e = blockIdx.x * blockDim.x + threadIdx.x;
  if (e >= NEDGE) return;
  int r = row[e], c = col[e];
  float nr = dinv[r] * dinv[c];
  const float* xs = x + (long long)r * Cs;
  float* op = out + (long long)c * C;
  for (int k = 0; k < C; ++k) atomicAdd(&op[k], nr * xs[k]);
}

// ---------------- final projection ----------------
__global__ void k_final(const _Float16* __restrict__ h, const float* __restrict__ Wo,
                        const float* __restrict__ bo, float* __restrict__ out) {
  int t = blockIdx.x * blockDim.x + threadIdx.x;
  if (t >= NTOTC) return;
  float s = bo[0];
  const _Float16* p = h + (long long)t * 256;
  for (int c = 0; c < 256; ++c) s += (float)p[c] * Wo[c];
  out[t] = s;
}

// =====================================================================
enum {
  IN_X = 0, IN_EDGE = 1,
  C1W0=2,  C1B0=3,  C1G0=4,  C1T0=5,
  C1W1=6,  C1B1=7,  C1G1=8,  C1T1=9,
  C1W2=10, C1B2=11, C1G2=12, C1T2=13,
  C2W0=14, C2B0=15, C2G0=16, C2T0=17,
  L1W=18, L1B=19, L1G=20, L1T=21,
  LGW=22, LGB=23, LGG=24, LGT=25,
  M1W=26, M1B=27, M1G=28, M1T=29,
  M2W=30, M2B=31, M2G=32, M2T=33,
  MOW=34, MOB=35,
  T1W0=36, T1W1=37, T1W2=38, T1W3=39, T1B=40,
  T2W0=41, T2W1=42, T2W2=43, T2W3=44, T2B=45,
};

extern "C" void kernel_launch(void* const* d_in, const int* in_sizes, int n_in,
                              void* d_out, int out_size, void* d_ws, size_t ws_size,
                              hipStream_t stream) {
  (void)in_sizes; (void)n_in; (void)out_size; (void)ws_size;
  const float* x4   = (const float*)d_in[IN_X];
  const int*   rowp = (const int*)d_in[IN_EDGE];
  const int*   colp = rowp + NEDGE;
  float* outp = (float*)d_out;

  // ---- workspace carve-up ----
  char* w = (char*)d_ws;
  size_t o = 0;
  auto take = [&](size_t bytes) -> char* {
    char* p = w + o; o = (o + bytes + 255) & ~(size_t)255; return p;
  };
  float*    YBIG  = (float*)   take((size_t)ME * 128 * 4);   // f32 GEMM out / gram
  _Float16* XE    = (_Float16*)take((size_t)ME * 128 * 2);   // edge activations ping
  _Float16* H16   = (_Float16*)take((size_t)ME * 128 * 2);   // edge activations pong
  _Float16* X116  = (_Float16*)take((size_t)NTOTC * 64 * 2);
  _Float16* X216  = (_Float16*)take((size_t)NTOTC * 128 * 2);
  int*      IDX   = (int*)     take((size_t)ME * 4);
  float*    NORMS = (float*)   take((size_t)NTOTC * 4);
  float*    STATS = (float*)   take((size_t)2 * 1024 * 4);
  _Float16* POS16 = (_Float16*)take((size_t)NTOTC * 3 * 2);
  _Float16* XPAD  = (_Float16*)take((size_t)NTOTC * 64 * 2);
  float*    DEG   = (float*)   take((size_t)NTOTC * 4);
  float*    DINV  = (float*)   take((size_t)NTOTC * 4);
  float*    XKA   = (float*)   take((size_t)NTOTC * 128 * 4);
  float*    XKB   = (float*)   take((size_t)NTOTC * 128 * 4);
  float*    OUTT  = (float*)   take((size_t)NTOTC * 128 * 4);
  float*    W1F   = (float*)   take((size_t)NTOTC * 64 * 4);
  _Float16* W116  = (_Float16*)take((size_t)NTOTC * 64 * 2);
  _Float16* W216  = (_Float16*)take((size_t)NTOTC * 128 * 2);
  _Float16* CC192 = (_Float16*)take((size_t)NTOTC * 192 * 2);
  _Float16* CC1K  = (_Float16*)take((size_t)NTOTC * 1024 * 2);
  _Float16* O116  = (_Float16*)take((size_t)NTOTC * 512 * 2);
  _Float16* O216  = (_Float16*)take((size_t)NTOTC * 512 * 2);
  _Float16* H116  = (_Float16*)take((size_t)NTOTC * 512 * 2);
  _Float16* H216  = (_Float16*)take((size_t)NTOTC * 256 * 2);
  // f16 transposed weights (N x Kpad)
  _Float16* Wc1l0 = (_Float16*)take(64  * 32   * 2);
  _Float16* Wc1l1 = (_Float16*)take(64  * 64   * 2);
  _Float16* Wc1l2 = (_Float16*)take(64  * 64   * 2);
  _Float16* Wc2   = (_Float16*)take(128 * 128  * 2);
  _Float16* Wlin1 = (_Float16*)take(512 * 192  * 2);
  _Float16* Wling = (_Float16*)take(512 * 192  * 2);
  _Float16* Wmix1 = (_Float16*)take(512 * 1024 * 2);
  _Float16* Wmix2 = (_Float16*)take(256 * 512  * 2);
  _Float16* Wt1[4]; for (int i = 0; i < 4; ++i) Wt1[i] = (_Float16*)take(64  * 32 * 2);
  _Float16* Wt2[4]; for (int i = 0; i < 4; ++i) Wt2[i] = (_Float16*)take(128 * 64 * 2);

  // ---- helpers ----
  auto gemm = [&](const _Float16* A, long long sA, const _Float16* Bt, long long sB,
                  const float* bias, float* C, long long sC,
                  int M, int Nn, int Kd, int flags, float* stats, int nb) {
    dim3 grid((M + 63) / 64, Nn / 64, nb);           // wave = 16x64 strip, block = 64 rows
    k_gemm<<<grid, dim3(128, 1, 1), 0, stream>>>(A, sA, Bt, sB, bias, C, sC,
                                                 M, Nn, Kd, flags, stats);
  };
  auto wt = [&](int pi, _Float16* dst, int K, int N, int Kpad) {
    int tot = N * Kpad;
    k_wt<<<(tot + 255) / 256, 256, 0, stream>>>((const float*)d_in[pi], dst, K, N, Kpad);
  };
  auto zstats = [&](int Nc) { hipMemsetAsync(STATS, 0, (size_t)2 * Nc * 4, stream); };
  auto bnfin = [&](int Nc, int gi, int ti, float invM) {
    k_bn_finalize<<<(Nc + 63) / 64, 64, 0, stream>>>(STATS, (const float*)d_in[gi],
                                                     (const float*)d_in[ti], Nc, invM);
  };
  auto bnapp = [&](const float* Y, _Float16* dst, long long total, int Nc) {
    k_bn_apply<<<(unsigned)((total + 255) / 256), 256, 0, stream>>>(Y, STATS, dst, total, Nc);
  };

  // ---- weight prep ----
  wt(C1W0, Wc1l0, 6, 64, 32);
  wt(C1W1, Wc1l1, 64, 64, 64);
  wt(C1W2, Wc1l2, 64, 64, 64);
  wt(C2W0, Wc2, 128, 128, 128);
  wt(L1W, Wlin1, 192, 512, 192);
  wt(LGW, Wling, 192, 512, 192);
  wt(M1W, Wmix1, 1024, 512, 1024);
  wt(M2W, Wmix2, 512, 256, 512);
  for (int i = 0; i < 4; ++i) wt(T1W0 + i, Wt1[i], 4, 64, 32);
  for (int i = 0; i < 4; ++i) wt(T2W0 + i, Wt2[i], 64, 128, 64);

  // ================= EdgeConv 1 =================
  k_pos16<<<NTOTC / 256, 256, 0, stream>>>(x4, POS16);
  k_knn_pos<<<NTOTC / 256, 256, 0, stream>>>(x4, IDX);
  k_edge_features<<<ME / 256, 256, 0, stream>>>(POS16, IDX, XE, 3, 32);
  const float invME = 1.0f / (float)ME;
  // L0: 32 -> 64
  zstats(64);
  gemm(XE, 0, Wc1l0, 0, (const float*)d_in[C1B0], YBIG, 0, ME, 64, 32, 1, STATS, 1);
  bnfin(64, C1G0, C1T0, invME);
  bnapp(YBIG, H16, (long long)ME * 64, 64);
  // L1: 64 -> 64
  zstats(64);
  gemm(H16, 0, Wc1l1, 0, (const float*)d_in[C1B1], YBIG, 0, ME, 64, 64, 1, STATS, 1);
  bnfin(64, C1G1, C1T1, invME);
  bnapp(YBIG, XE, (long long)ME * 64, 64);
  // L2: 64 -> 64
  zstats(64);
  gemm(XE, 0, Wc1l2, 0, (const float*)d_in[C1B2], YBIG, 0, ME, 64, 64, 1, STATS, 1);
  bnfin(64, C1G2, C1T2, invME);
  bnapp(YBIG, H16, (long long)ME * 64, 64);
  k_max_k<<<(NTOTC * 64) / 256, 256, 0, stream>>>(H16, X116, 64);

  // ================= KNN in 64-d (WMMA gram) =================
  k_norms<<<NTOTC / 256, 256, 0, stream>>>(X116, NORMS);
  gemm(X116, (long long)NPTS * 64, X116, (long long)NPTS * 64, nullptr,
       YBIG, (long long)NPTS * NPTS, NPTS, NPTS, 64, 0, nullptr, BSZ);
  k_topk_gram<<<NTOTC / 256, 256, 0, stream>>>(YBIG, NORMS, IDX);

  // ================= EdgeConv 2 =================
  k_edge_features<<<ME / 256, 256, 0, stream>>>(X116, IDX, XE, 64, 128);
  zstats(128);
  gemm(XE, 0, Wc2, 0, (const float*)d_in[C2B0], YBIG, 0, ME, 128, 128, 1, STATS, 1);
  bnfin(128, C2G0, C2T0, invME);
  bnapp(YBIG, H16, (long long)ME * 128, 128);
  k_max_k<<<(NTOTC * 128) / 256, 256, 0, stream>>>(H16, X216, 128);

  // ================= lin1 branch =================
  const float invNT = 1.0f / (float)NTOTC;
  k_concat2<<<(unsigned)(((long long)NTOTC * 192 + 255) / 256), 256, 0, stream>>>(
      X116, 64, X216, 128, CC192, NTOTC);
  zstats(512);
  gemm(CC192, 0, Wlin1, 0, (const float*)d_in[L1B], YBIG, 0, NTOTC, 512, 192, 1, STATS, 1);
  bnfin(512, L1G, L1T, invNT);
  bnapp(YBIG, O116, (long long)NTOTC * 512, 512);

  // ================= TAG convs =================
  hipMemsetAsync(DEG, 0, (size_t)NTOTC * 4, stream);
  k_deg<<<NEDGE / 256, 256, 0, stream>>>(colp, DEG);
  k_dinv<<<NTOTC / 256, 256, 0, stream>>>(DEG, DINV);

  // ---- tag1: x_input(4) -> 64, 3 hops ----
  k_cast_pad<<<(unsigned)(((long long)NTOTC * 32 + 255) / 256), 256, 0, stream>>>(
      x4, 4, 4, XPAD, 32, (long long)NTOTC * 32);
  gemm(XPAD, 0, Wt1[0], 0, nullptr, OUTT, 0, NTOTC, 64, 32, 0, nullptr, 1);
  {
    const float* cur = x4; int curCs = 4;
    float* bufs[2] = { XKA, XKB };
    for (int hop = 1; hop <= 3; ++hop) {
      float* dst = bufs[hop & 1];
      hipMemsetAsync(dst, 0, (size_t)NTOTC * 4 * 4, stream);
      k_prop<<<NEDGE / 256, 256, 0, stream>>>(cur, rowp, colp, DINV, dst, 4, curCs);
      k_cast_pad<<<(unsigned)(((long long)NTOTC * 32 + 255) / 256), 256, 0, stream>>>(
          dst, 4, 4, XPAD, 32, (long long)NTOTC * 32);
      gemm(XPAD, 0, Wt1[hop], 0, nullptr, OUTT, 0, NTOTC, 64, 32, 2, nullptr, 1);
      cur = dst; curCs = 4;
    }
  }
  k_brc<<<(unsigned)(((long long)NTOTC * 64 + 255) / 256), 256, 0, stream>>>(
      OUTT, (const float*)d_in[T1B], W1F, W116, (long long)NTOTC * 64, 64);

  // ---- tag2: w1(64) -> 128, 3 hops ----
  k_cast_pad<<<(unsigned)(((long long)NTOTC * 64 + 255) / 256), 256, 0, stream>>>(
      W1F, 64, 64, XPAD, 64, (long long)NTOTC * 64);
  gemm(XPAD, 0, Wt2[0], 0, nullptr, OUTT, 0, NTOTC, 128, 64, 0, nullptr, 1);
  {
    const float* cur = W1F;
    float* bufs[2] = { XKA, XKB };
    for (int hop = 1; hop <= 3; ++hop) {
      float* dst = bufs[hop & 1];
      hipMemsetAsync(dst, 0, (size_t)NTOTC * 64 * 4, stream);
      k_prop<<<NEDGE / 256, 256, 0, stream>>>(cur, rowp, colp, DINV, dst, 64, 64);
      k_cast_pad<<<(unsigned)(((long long)NTOTC * 64 + 255) / 256), 256, 0, stream>>>(
          dst, 64, 64, XPAD, 64, (long long)NTOTC * 64);
      gemm(XPAD, 0, Wt2[hop], 0, nullptr, OUTT, 0, NTOTC, 128, 64, 2, nullptr, 1);
      cur = dst;
    }
  }
  k_brc<<<(unsigned)(((long long)NTOTC * 128 + 255) / 256), 256, 0, stream>>>(
      OUTT, (const float*)d_in[T2B], nullptr, W216, (long long)NTOTC * 128, 128);

  // ---- lin_gcnn1 ----
  k_concat2<<<(unsigned)(((long long)NTOTC * 192 + 255) / 256), 256, 0, stream>>>(
      W116, 64, W216, 128, CC192, NTOTC);
  zstats(512);
  gemm(CC192, 0, Wling, 0, (const float*)d_in[LGB], YBIG, 0, NTOTC, 512, 192, 1, STATS, 1);
  bnfin(512, LGG, LGT, invNT);
  bnapp(YBIG, O216, (long long)NTOTC * 512, 512);

  // ================= mix head =================
  k_concat2<<<(unsigned)(((long long)NTOTC * 1024 + 255) / 256), 256, 0, stream>>>(
      O116, 512, O216, 512, CC1K, NTOTC);
  zstats(512);
  gemm(CC1K, 0, Wmix1, 0, (const float*)d_in[M1B], YBIG, 0, NTOTC, 512, 1024, 1, STATS, 1);
  bnfin(512, M1G, M1T, invNT);
  bnapp(YBIG, H116, (long long)NTOTC * 512, 512);

  zstats(256);
  gemm(H116, 0, Wmix2, 0, (const float*)d_in[M2B], YBIG, 0, NTOTC, 256, 512, 1, STATS, 1);
  bnfin(256, M2G, M2T, invNT);
  bnapp(YBIG, H216, (long long)NTOTC * 256, 256);

  k_final<<<NTOTC / 256, 256, 0, stream>>>(H216, (const float*)d_in[MOW],
                                           (const float*)d_in[MOB], outp);
}
